// TemporalDynamicsModule_66924180407364
// MI455X (gfx1250) — matrix operand order repaired and checked by reading, not compiled, not run on hardware
//
#include <hip/hip_runtime.h>
#include <math.h>

// ---------------------------------------------------------------------------
// TemporalDynamicsModule forward for MI455X (gfx1250, wave32, WMMA, async LDS).
// All heavy GEMMs: C = act(alpha * A(MxK) @ B^T(NxK) + bias) + residual
// A and B are f16 in global memory (weights pre-converted once, activations
// produced in f16 by the producing kernels). Tiles are staged into LDS with
// GLOBAL_LOAD_ASYNC_TO_LDS_B128 (ASYNCcnt), double buffered, so the inner
// loop is ds_load_b128 fragments + v_wmma_f32_16x16x32_f16 only.
// ---------------------------------------------------------------------------

#define Lz   6
#define Bz   2
#define Sz   1024
#define Dz   1024
#define Hz   16
#define HDz  64
#define HIDz 2048
#define FFz  4096
#define BSD  (Bz * Sz * Dz)   // 2097152
#define M2   (Bz * Sz)        // 2048

typedef __attribute__((ext_vector_type(16))) _Float16 v16h;
typedef __attribute__((ext_vector_type(8)))  _Float16 v8h;
typedef __attribute__((ext_vector_type(8)))  float    v8f;

union AFrag { v16h v; v8h h[2]; };

__device__ __forceinline__ float gelu_f(float x) {
  return 0.5f * x * (1.0f + erff(x * 0.70710678118654752440f));
}

// ---------------------------------------------------------------------------
// WMMA GEMM with async-LDS staging.
//  - workgroup 256 threads = 8 waves; macro tile 128(M) x 64(N), K-step 32
//  - each wave owns a 32x32 C tile -> 4 v_wmma_f32_16x16x32_f16 per K-step
//  - LDS double buffer: per buffer A 128x32 f16 (8KB) + B 64x32 f16 (4KB)
//  - stage-in: per thread 3x global_load_async_to_lds_b128 per tile
//    (A: 2 chunks, B: 1 chunk); s_wait_asynccnt 3 keeps next tile in flight
//  - OOB rows/cols are clamped (not predicated) so EXEC stays all-ones
// Fragment layouts per CDNA5 ISA 7.12.2:
//   A 16x32 f16 : lane<16 -> M=lane, K {kg*8..+7, 16+kg*8..+7}, kg=lane>>4
//   B 32x16 f16 : lane    -> N=lane&15, K {kg*8..+7, 16+kg*8..+7}, kg=lane>>4
//   C 16x16 f32 : vgpr g, lane<16 -> M=g,N=lane ; lane>=16 -> M=g+8,N=lane-16
// ---------------------------------------------------------------------------
__global__ __launch_bounds__(256)
void gemm_wmma_kernel(const _Float16* __restrict__ A, const _Float16* __restrict__ Bw,
                      const float* __restrict__ bias, const float* __restrict__ Res,
                      float* __restrict__ C32, _Float16* __restrict__ C16,
                      int M, int N, int K,
                      long long lda, long long ldb, long long ldc,
                      long long sA, long long sB, long long sC,
                      int act, float alpha)
{
  // two buffers, each: A tile 4096 halves, B tile 2048 halves
  __shared__ __align__(128) _Float16 lds[2 * 6144];

  A  += (long long)blockIdx.z * sA;
  Bw += (long long)blockIdx.z * sB;

  const int tid  = threadIdx.x;
  const int lane = tid & 31;
  const int wave = tid >> 5;
  const int wr   = wave >> 1;     // 0..3 : 32-row slice
  const int wc   = wave & 1;      // 0..1 : 32-col slice
  const int m0   = blockIdx.y * 128;
  const int n0   = blockIdx.x * 64;
  const int mrow = lane & 15;
  const int kg   = lane >> 4;

  const unsigned ldsBase = (unsigned)(uintptr_t)(&lds[0]);   // LDS byte offset

  auto issue_tile = [&](int kt, int buf) {
    const int kBase = kt << 5;
    const unsigned ldsA = ldsBase + (unsigned)(buf * 6144 * 2);
    const unsigned ldsB = ldsA + 8192u;
    // A tile: 512 b128 chunks (row,seg), 2 per thread
#pragma unroll
    for (int c = 0; c < 2; ++c) {
      int chunk = tid + c * 256;
      int row = chunk >> 2;
      int seg = chunk & 3;
      int gm = m0 + row; if (gm >= M) gm = M - 1;          // clamp, keep EXEC full
      const _Float16* src = A + (long long)gm * lda + kBase + seg * 8;
      unsigned loff = ldsA + (unsigned)((row * 32 + seg * 8) * 2);
      asm volatile("global_load_async_to_lds_b128 %0, %1, off"
                   :: "v"(loff), "v"((unsigned long long)(uintptr_t)src)
                   : "memory");
    }
    // B tile: 256 b128 chunks, 1 per thread
    {
      int row = tid >> 2;
      int seg = tid & 3;
      int gn = n0 + row; if (gn >= N) gn = N - 1;
      const _Float16* src = Bw + (long long)gn * ldb + kBase + seg * 8;
      unsigned loff = ldsB + (unsigned)((row * 32 + seg * 8) * 2);
      asm volatile("global_load_async_to_lds_b128 %0, %1, off"
                   :: "v"(loff), "v"((unsigned long long)(uintptr_t)src)
                   : "memory");
    }
  };

  v8f acc[2][2];
#pragma unroll
  for (int i = 0; i < 2; ++i)
#pragma unroll
    for (int j = 0; j < 2; ++j)
#pragma unroll
      for (int e = 0; e < 8; ++e) acc[i][j][e] = 0.0f;

  const int kT = K >> 5;          // K is always a multiple of 32 here
  issue_tile(0, 0);

  for (int kt = 0; kt < kT; ++kt) {
    const int cur = kt & 1;
    if (kt + 1 < kT) {
      issue_tile(kt + 1, cur ^ 1);                 // next tile in flight
      asm volatile("s_wait_asynccnt 0x3" ::: "memory");   // tile kt complete
    } else {
      asm volatile("s_wait_asynccnt 0x0" ::: "memory");
    }
    __syncthreads();

    const _Float16* Ab = &lds[cur * 6144];
    const _Float16* Bb = &lds[cur * 6144 + 4096];

    AFrag a0, a1;
    {
      const _Float16* p0 = Ab + (wr * 32 + mrow) * 32;
      a0.h[0] = *(const v8h*)(p0 + kg * 8);
      a0.h[1] = *(const v8h*)(p0 + 16 + kg * 8);
      const _Float16* p1 = Ab + (wr * 32 + 16 + mrow) * 32;
      a1.h[0] = *(const v8h*)(p1 + kg * 8);
      a1.h[1] = *(const v8h*)(p1 + 16 + kg * 8);
    }
    AFrag bf0, bf1;
    bf0.h[0] = *(const v8h*)(Bb + (wc * 32 + mrow) * 32 + kg * 8);
    bf0.h[1] = *(const v8h*)(Bb + (wc * 32 + mrow) * 32 + 16 + kg * 8);
    bf1.h[0] = *(const v8h*)(Bb + (wc * 32 + 16 + mrow) * 32 + kg * 8);
    bf1.h[1] = *(const v8h*)(Bb + (wc * 32 + 16 + mrow) * 32 + 16 + kg * 8);

    acc[0][0] = __builtin_amdgcn_wmma_f32_16x16x32_f16(false, a0.v, false, bf0.v, (short)0, acc[0][0], false, false);
    acc[0][1] = __builtin_amdgcn_wmma_f32_16x16x32_f16(false, a0.v, false, bf1.v, (short)0, acc[0][1], false, false);
    acc[1][0] = __builtin_amdgcn_wmma_f32_16x16x32_f16(false, a1.v, false, bf0.v, (short)0, acc[1][0], false, false);
    acc[1][1] = __builtin_amdgcn_wmma_f32_16x16x32_f16(false, a1.v, false, bf1.v, (short)0, acc[1][1], false, false);

    __syncthreads();    // all waves done reading buf before it is refilled
  }

  // ---- epilogue: alpha, bias, activation, residual, f32/f16 stores ----
#pragma unroll
  for (int i = 0; i < 2; ++i) {
#pragma unroll
    for (int j = 0; j < 2; ++j) {
      const int n = n0 + wc * 32 + j * 16 + mrow;
#pragma unroll
      for (int g = 0; g < 8; ++g) {
        const int m = m0 + wr * 32 + i * 16 + kg * 8 + g;
        if (m < M && n < N) {
          float v = acc[i][j][g] * alpha;
          if (bias) v += bias[n];
          if      (act == 1) v = gelu_f(v);
          else if (act == 2) v = 1.0f / (1.0f + expf(-v));
          else if (act == 3) v = expf(v);
          if (Res) v += Res[(long long)m * ldc + n];
          const long long o = (long long)blockIdx.z * sC + (long long)m * ldc + n;
          if (C32) C32[o] = v;
          if (C16) C16[o] = (_Float16)v;
        }
      }
    }
  }
}

// ---------------------------------------------------------------------------
// Elementwise / reduction helpers
// ---------------------------------------------------------------------------
__global__ void cvt16_kernel(const float* __restrict__ src, _Float16* __restrict__ dst, int n)
{
  int i = blockIdx.x * 256 + threadIdx.x;
  if (i < n) dst[i] = (_Float16)src[i];
}

__global__ void add_pos_kernel(const float* __restrict__ x, const float* __restrict__ pos,
                               float* __restrict__ out, int n, int sd)
{
  int i = blockIdx.x * 256 + threadIdx.x;
  if (i < n) out[i] = x[i] + pos[i % sd];
}

// LayerNorm; writes f32 and/or f16 outputs, optional fused exact GELU.
__global__ void layernorm_kernel(const float* __restrict__ in,
                                 float* __restrict__ out32, _Float16* __restrict__ out16,
                                 const float* __restrict__ g, const float* __restrict__ b,
                                 int Dn, int act)
{
  __shared__ float red[256];
  __shared__ float red2[256];
  const int tid = threadIdx.x;
  const long long row = blockIdx.x;
  const float* xr = in + row * Dn;
  float s = 0.f, s2 = 0.f;
  for (int j = tid; j < Dn; j += 256) { float v = xr[j]; s += v; s2 += v * v; }
  red[tid] = s; red2[tid] = s2; __syncthreads();
  for (int st = 128; st > 0; st >>= 1) {
    if (tid < st) { red[tid] += red[tid + st]; red2[tid] += red2[tid + st]; }
    __syncthreads();
  }
  float mean = red[0] / (float)Dn;
  float var  = red2[0] / (float)Dn - mean * mean;
  float inv  = rsqrtf(var + 1e-5f);
  for (int j = tid; j < Dn; j += 256) {
    float v = (xr[j] - mean) * inv * g[j] + b[j];
    if (act) v = gelu_f(v);
    if (out32) out32[row * Dn + j] = v;
    if (out16) out16[row * Dn + j] = (_Float16)v;
  }
}

// conv_w[i] (O, I, 3) -> f16 wp[O][k*D + I] so conv becomes GEMM with K=3D
__global__ void pack_convw_kernel(const float* __restrict__ w, _Float16* __restrict__ wp, int n)
{
  int i = blockIdx.x * 256 + threadIdx.x;
  if (i >= n) return;
  int c  = i % Dz;
  int t  = i / Dz;
  int kk = t % 3;
  int o  = t / 3;
  wp[i] = (_Float16)w[((long long)o * Dz + c) * 3 + kk];
}

// A_conv[b][t][kk*D + c] = h[b][t + (kk-1)*dil][c]  (zero padded), f16 -> f16
__global__ void conv_gather_kernel(const _Float16* __restrict__ h, _Float16* __restrict__ a,
                                   int dil, int n)
{
  int i = blockIdx.x * 256 + threadIdx.x;
  if (i >= n) return;
  int c  = i % Dz;
  int t2 = i / Dz;
  int kk = t2 % 3;
  t2 /= 3;
  int t  = t2 % Sz;
  int b  = t2 / Sz;
  int tt = t + (kk - 1) * dil;
  _Float16 v = (_Float16)0.f;
  if (tt >= 0 && tt < Sz) v = h[((long long)b * Sz + tt) * Dz + c];
  a[i] = v;
}

// qkv f16 (B,S,3,H,HD) -> q,k (B,H,S,HD) with RoPE, v^T (B,H,HD,S)
__global__ void qkv_split_rope_kernel(const _Float16* __restrict__ qkv,
                                      _Float16* __restrict__ q, _Float16* __restrict__ k,
                                      _Float16* __restrict__ vT, int n)
{
  int i = blockIdx.x * 256 + threadIdx.x;
  if (i >= n) return;
  int d = i % HDz;
  int t = i / HDz;
  int s = t % Sz; t /= Sz;
  int h = t % Hz;
  int b = t / Hz;
  long long base = (((long long)(b * Sz + s) * 3) * Hz + h) * HDz;
  float qv = (float)qkv[base + d];
  float kv = (float)qkv[base + (long long)Hz * HDz + d];
  float vv = (float)qkv[base + 2LL * Hz * HDz + d];
  const int half = HDz / 2;
  int j = (d < half) ? d : d - half;
  float freq = (float)s * powf(10000.f, -2.f * (float)j / (float)HDz);
  float c = cosf(freq), sn = sinf(freq);
  float qo, ko;
  if (d < half) {
    float q2 = (float)qkv[base + d + half];
    float k2 = (float)qkv[base + (long long)Hz * HDz + d + half];
    qo = qv * c - q2 * sn;  ko = kv * c - k2 * sn;
  } else {
    float q1 = (float)qkv[base + d - half];
    float k1 = (float)qkv[base + (long long)Hz * HDz + d - half];
    qo = qv * c + q1 * sn;  ko = kv * c + k1 * sn;
  }
  q[i] = (_Float16)qo;
  k[i] = (_Float16)ko;
  vT[(((long long)b * Hz + h) * HDz + d) * Sz + s] = (_Float16)vv;
}

// causal softmax over f32 logits, writes f16 probabilities
__global__ void softmax_causal_kernel(const float* __restrict__ sc, _Float16* __restrict__ pr)
{
  __shared__ float red[256];
  const int tid = threadIdx.x;
  const long long row = blockIdx.x;        // over B*H*S rows
  const int q = (int)(row % Sz);
  const float* r = sc + row * (long long)Sz;
  _Float16* o = pr + row * (long long)Sz;
  float mx = -3.402823e38f;
  for (int j = tid; j <= q; j += 256) mx = fmaxf(mx, r[j]);
  red[tid] = mx; __syncthreads();
  for (int st = 128; st > 0; st >>= 1) { if (tid < st) red[tid] = fmaxf(red[tid], red[tid + st]); __syncthreads(); }
  mx = red[0]; __syncthreads();
  float sum = 0.f;
  for (int j = tid; j <= q; j += 256) sum += expf(r[j] - mx);
  red[tid] = sum; __syncthreads();
  for (int st = 128; st > 0; st >>= 1) { if (tid < st) red[tid] += red[tid + st]; __syncthreads(); }
  float inv = 1.0f / red[0];
  for (int j = tid; j < Sz; j += 256)
    o[j] = (_Float16)((j <= q) ? expf(r[j] - mx) * inv : 0.0f);
}

// (B,H,S,HD) f16 -> (B,S,H*HD) f16
__global__ void merge_heads_kernel(const _Float16* __restrict__ t, _Float16* __restrict__ out, int n)
{
  int i = blockIdx.x * 256 + threadIdx.x;
  if (i >= n) return;
  int d = i % HDz;
  int r = i / HDz;
  int h = r % Hz; r /= Hz;
  int s = r % Sz;
  int b = r / Sz;
  out[i] = t[(((long long)b * Hz + h) * Sz + s) * HDz + d];
}

// concat two f32 (rows x D) into f16 (rows x 2D)
__global__ void concat2_kernel(const float* __restrict__ a, const float* __restrict__ b2,
                               _Float16* __restrict__ out, int n)
{
  int i = blockIdx.x * 256 + threadIdx.x;
  if (i >= n) return;
  int c = i % (2 * Dz);
  int r = i / (2 * Dz);
  float v = (c < Dz) ? a[(long long)r * Dz + c] : b2[(long long)r * Dz + (c - Dz)];
  out[i] = (_Float16)v;
}

__global__ void gate_combine_kernel(float* __restrict__ x, const float* __restrict__ g,
                                    const float* __restrict__ conv, const float* __restrict__ attn,
                                    int n)
{
  int i = blockIdx.x * 256 + threadIdx.x;
  if (i < n) x[i] = x[i] + g[i] * conv[i] + (1.0f - g[i]) * attn[i];
}

__global__ void gather_last_kernel(const float* __restrict__ src,
                                   float* __restrict__ dst32, _Float16* __restrict__ dst16, int n)
{
  int i = blockIdx.x * 256 + threadIdx.x;
  if (i >= n) return;
  int d = i % Dz;
  int b = i / Dz;
  float v = src[((long long)b * Sz + (Sz - 1)) * Dz + d];
  if (dst32) dst32[i] = v;
  if (dst16) dst16[i] = (_Float16)v;
}

// u16[row][0..HID) = relu(mean(comb[row][0..2*HID)))
__global__ void uin_kernel(const float* __restrict__ comb, _Float16* __restrict__ u)
{
  __shared__ float red[256];
  const int tid = threadIdx.x;
  const int row = blockIdx.x;
  const float* r = comb + (long long)row * (2 * HIDz);
  float s = 0.f;
  for (int j = tid; j < 2 * HIDz; j += 256) s += r[j];
  red[tid] = s; __syncthreads();
  for (int st = 128; st > 0; st >>= 1) { if (tid < st) red[tid] += red[tid + st]; __syncthreads(); }
  float val = fmaxf(red[0] / (float)(2 * HIDz), 0.0f);
  for (int j = tid; j < HIDz; j += 256) u[(long long)row * HIDz + j] = (_Float16)val;
}

__global__ void add2_kernel(const float* __restrict__ a, const float* __restrict__ b2,
                            float* __restrict__ out, int n)
{
  int i = blockIdx.x * 256 + threadIdx.x;
  if (i < n) out[i] = a[i] + b2[i];
}

// ---------------------------------------------------------------------------
// Host side
// ---------------------------------------------------------------------------
static inline void run_gemm(hipStream_t st, const _Float16* A, const _Float16* Bw,
                            const float* bias, const float* Res,
                            float* C32, _Float16* C16,
                            int M, int N, int K,
                            long long lda, long long ldb, long long ldc,
                            int batch, long long sA, long long sB, long long sC,
                            int act, float alpha)
{
  dim3 grid((N + 63) / 64, (M + 127) / 128, batch);
  gemm_wmma_kernel<<<grid, dim3(256), 0, st>>>(A, Bw, bias, Res, C32, C16, M, N, K,
                                               lda, ldb, ldc, sA, sB, sC, act, alpha);
}

#define EWG(n) dim3(((n) + 255) / 256)

extern "C" void kernel_launch(void* const* d_in, const int* in_sizes, int n_in,
                              void* d_out, int out_size, void* d_ws, size_t ws_size,
                              hipStream_t stream)
{
  (void)in_sizes; (void)n_in; (void)out_size; (void)ws_size;

  const float* x_in     = (const float*)d_in[0];
  const float* actions  = (const float*)d_in[1];
  const float* pos_emb  = (const float*)d_in[2];
  const float* conv_w   = (const float*)d_in[3];
  const float* conv_b   = (const float*)d_in[4];
  const float* convln_g = (const float*)d_in[5];
  const float* convln_b = (const float*)d_in[6];
  const float* ln1_g    = (const float*)d_in[7];
  const float* ln1_b    = (const float*)d_in[8];
  const float* ln2_g    = (const float*)d_in[9];
  const float* ln2_b    = (const float*)d_in[10];
  const float* ln3_g    = (const float*)d_in[11];
  const float* ln3_b    = (const float*)d_in[12];
  const float* qkv_w    = (const float*)d_in[13];
  const float* qkv_b    = (const float*)d_in[14];
  const float* ao_w     = (const float*)d_in[15];
  const float* ao_b     = (const float*)d_in[16];
  const float* gate_w   = (const float*)d_in[17];
  const float* gate_b   = (const float*)d_in[18];
  const float* ffn1_w   = (const float*)d_in[19];
  const float* ffn1_b   = (const float*)d_in[20];
  const float* ffn2_w   = (const float*)d_in[21];
  const float* ffn2_b   = (const float*)d_in[22];
  const float* on_g     = (const float*)d_in[23];
  const float* on_b     = (const float*)d_in[24];
  const float* op_w     = (const float*)d_in[25];
  const float* op_b     = (const float*)d_in[26];
  const float* se_w     = (const float*)d_in[27];
  const float* se_b     = (const float*)d_in[28];
  const float* ae_w     = (const float*)d_in[29];
  const float* ae_b     = (const float*)d_in[30];
  const float* t0_w     = (const float*)d_in[31];
  const float* t0_b     = (const float*)d_in[32];
  const float* t0ln_g   = (const float*)d_in[33];
  const float* t0ln_b   = (const float*)d_in[34];
  const float* t1_w     = (const float*)d_in[35];
  const float* t1_b     = (const float*)d_in[36];
  const float* t1ln_g   = (const float*)d_in[37];
  const float* t1ln_b   = (const float*)d_in[38];
  const float* t2_w     = (const float*)d_in[39];
  const float* t2_b     = (const float*)d_in[40];
  const float* unc_w    = (const float*)d_in[41];
  const float* unc_b    = (const float*)d_in[42];

  // ---- f32 workspace ----
  float* p = (float*)d_ws;
  float* xbuf    = p; p += BSD;                            // running residual stream
  float* convbuf = p; p += BSD;                            // conv branch (f32)
  float* attnf   = p; p += BSD;                            // attn branch (f32)
  float* gatebuf = p; p += BSD;                            // sigmoid gate
  float* scores  = p; p += (long long)Bz * Hz * Sz * Sz;   // attention logits
  float* comb    = p; p += Bz * 2 * HIDz;
  float* h0      = p; p += Bz * HIDz;
  float* h2      = p; p += Bz * HIDz;
  float* csb     = p; p += Bz * Dz;

  // ---- f16 workspace (all carve sizes are multiples of 2048 halves) ----
  _Float16* q16 = (_Float16*)p;
  _Float16* qkvw16  = q16; q16 += (long long)Lz * 3 * Dz * Dz;
  _Float16* aow16   = q16; q16 += (long long)Lz * Dz * Dz;
  _Float16* gatew16 = q16; q16 += (long long)Lz * Dz * 2 * Dz;
  _Float16* ffn1w16 = q16; q16 += (long long)Lz * FFz * Dz;
  _Float16* ffn2w16 = q16; q16 += (long long)Lz * Dz * FFz;
  _Float16* opw16   = q16; q16 += (long long)Dz * Dz;
  _Float16* sew16   = q16; q16 += (long long)HIDz * Dz;
  _Float16* aew16   = q16; q16 += (long long)HIDz * Dz;
  _Float16* t0w16   = q16; q16 += (long long)HIDz * 2 * HIDz;
  _Float16* t1w16   = q16; q16 += (long long)HIDz * HIDz;
  _Float16* t2w16   = q16; q16 += (long long)Dz * HIDz;
  _Float16* uncw16  = q16; q16 += (long long)Dz * HIDz;
  _Float16* wpack16 = q16; q16 += (long long)Dz * 3 * Dz;  // per-layer conv pack
  _Float16* hbuf16  = q16; q16 += BSD;                     // LN outputs (GEMM A)
  _Float16* aconv16 = q16; q16 += 3 * BSD;
  _Float16* qkv16   = q16; q16 += 3 * BSD;
  _Float16* qb16    = q16; q16 += BSD;                     // (B,H,S,HD)
  _Float16* kb16    = q16; q16 += BSD;
  _Float16* vT16    = q16; q16 += BSD;                     // (B,H,HD,S)
  _Float16* p16     = q16; q16 += (long long)Bz * Hz * Sz * Sz;  // probs
  _Float16* av16    = q16; q16 += BSD;                     // (B,H,S,HD)
  _Float16* attn16  = q16; q16 += BSD;                     // merged heads
  _Float16* cat16   = q16; q16 += 2 * BSD;
  _Float16* ff116   = q16; q16 += 4 * BSD;
  _Float16* xout16  = q16; q16 += BSD;
  _Float16* comb16  = q16; q16 += Bz * 2 * HIDz;
  _Float16* csb16   = q16; q16 += Bz * Dz;
  _Float16* actl16  = q16; q16 += Bz * Dz;
  _Float16* h116    = q16; q16 += Bz * HIDz;
  _Float16* h316    = q16; q16 += Bz * HIDz;
  _Float16* ub16    = q16; q16 += Bz * HIDz;

  float* out       = (float*)d_out;
  float* dyn_out   = out;                 // (B,S,D)
  float* x_out     = out + BSD;           // (B,S,D) post-final-LN x
  float* ns_out    = x_out + BSD;         // (B,D)
  float* unc_out   = ns_out + Bz * Dz;    // (B,D)
  float* delta_out = unc_out + Bz * Dz;   // (B,D)

  const float scale = 0.125f;             // 1/sqrt(HD=64)

  // ---- one-time weight f32 -> f16 conversions ----
  cvt16_kernel<<<EWG(Lz*3*Dz*Dz), 256, 0, stream>>>(qkv_w,  qkvw16,  Lz*3*Dz*Dz);
  cvt16_kernel<<<EWG(Lz*Dz*Dz), 256, 0, stream>>>(ao_w,   aow16,   Lz*Dz*Dz);
  cvt16_kernel<<<EWG(Lz*Dz*2*Dz), 256, 0, stream>>>(gate_w, gatew16, Lz*Dz*2*Dz);
  cvt16_kernel<<<EWG(Lz*FFz*Dz), 256, 0, stream>>>(ffn1_w, ffn1w16, Lz*FFz*Dz);
  cvt16_kernel<<<EWG(Lz*Dz*FFz), 256, 0, stream>>>(ffn2_w, ffn2w16, Lz*Dz*FFz);
  cvt16_kernel<<<EWG(Dz*Dz), 256, 0, stream>>>(op_w,  opw16,  Dz*Dz);
  cvt16_kernel<<<EWG(HIDz*Dz), 256, 0, stream>>>(se_w,  sew16,  HIDz*Dz);
  cvt16_kernel<<<EWG(HIDz*Dz), 256, 0, stream>>>(ae_w,  aew16,  HIDz*Dz);
  cvt16_kernel<<<EWG(HIDz*2*HIDz), 256, 0, stream>>>(t0_w, t0w16, HIDz*2*HIDz);
  cvt16_kernel<<<EWG(HIDz*HIDz), 256, 0, stream>>>(t1_w, t1w16, HIDz*HIDz);
  cvt16_kernel<<<EWG(Dz*HIDz), 256, 0, stream>>>(t2_w,  t2w16,  Dz*HIDz);
  cvt16_kernel<<<EWG(Dz*HIDz), 256, 0, stream>>>(unc_w, uncw16, Dz*HIDz);

  // x = x + pos_emb[:, :S, :]
  add_pos_kernel<<<EWG(BSD), 256, 0, stream>>>(x_in, pos_emb, xbuf, BSD, Sz * Dz);

  for (int i = 0; i < Lz; ++i) {
    const int dil = 1 << i;

    // ---- conv branch: h = ln1(x); conv = gelu(convln(convW * h + b)) ----
    layernorm_kernel<<<M2, 256, 0, stream>>>(xbuf, nullptr, hbuf16, ln1_g + i * Dz, ln1_b + i * Dz, Dz, 0);
    pack_convw_kernel<<<EWG(Dz*3*Dz), 256, 0, stream>>>(conv_w + (long long)i * Dz * Dz * 3, wpack16, Dz * 3 * Dz);
    conv_gather_kernel<<<EWG(3 * BSD), 256, 0, stream>>>(hbuf16, aconv16, dil, 3 * BSD);
    run_gemm(stream, aconv16, wpack16, conv_b + i * Dz, nullptr, convbuf, nullptr,
             M2, Dz, 3 * Dz, 3 * Dz, 3 * Dz, Dz, 1, 0, 0, 0, 0, 1.0f);
    layernorm_kernel<<<M2, 256, 0, stream>>>(convbuf, convbuf, nullptr, convln_g + i * Dz, convln_b + i * Dz, Dz, 1);

    // ---- attention ----
    layernorm_kernel<<<M2, 256, 0, stream>>>(xbuf, nullptr, hbuf16, ln2_g + i * Dz, ln2_b + i * Dz, Dz, 0);
    run_gemm(stream, hbuf16, qkvw16 + (long long)i * 3 * Dz * Dz, qkv_b + i * 3 * Dz, nullptr, nullptr, qkv16,
             M2, 3 * Dz, Dz, Dz, Dz, 3 * Dz, 1, 0, 0, 0, 0, 1.0f);
    qkv_split_rope_kernel<<<EWG(BSD), 256, 0, stream>>>(qkv16, qb16, kb16, vT16, BSD);

    // scores[bh] = alpha * Q[bh] @ K[bh]^T     (M=N=S, K=HD)
    run_gemm(stream, qb16, kb16, nullptr, nullptr, scores, nullptr,
             Sz, Sz, HDz, HDz, HDz, Sz,
             Bz * Hz, (long long)Sz * HDz, (long long)Sz * HDz, (long long)Sz * Sz,
             0, scale);
    softmax_causal_kernel<<<Bz * Hz * Sz, 256, 0, stream>>>(scores, p16);

    // av[bh] = P[bh] @ (V^T[bh])^T             (M=S, N=HD, K=S)
    run_gemm(stream, p16, vT16, nullptr, nullptr, nullptr, av16,
             Sz, HDz, Sz, Sz, Sz, HDz,
             Bz * Hz, (long long)Sz * Sz, (long long)HDz * Sz, (long long)Sz * HDz,
             0, 1.0f);
    merge_heads_kernel<<<EWG(BSD), 256, 0, stream>>>(av16, attn16, BSD);
    run_gemm(stream, attn16, aow16 + (long long)i * Dz * Dz, ao_b + i * Dz, nullptr, attnf, nullptr,
             M2, Dz, Dz, Dz, Dz, Dz, 1, 0, 0, 0, 0, 1.0f);

    // ---- gate + combine ----
    concat2_kernel<<<EWG(2 * BSD), 256, 0, stream>>>(convbuf, attnf, cat16, 2 * BSD);
    run_gemm(stream, cat16, gatew16 + (long long)i * Dz * 2 * Dz, gate_b + i * Dz, nullptr, gatebuf, nullptr,
             M2, Dz, 2 * Dz, 2 * Dz, 2 * Dz, Dz, 1, 0, 0, 0, 2, 1.0f);
    gate_combine_kernel<<<EWG(BSD), 256, 0, stream>>>(xbuf, gatebuf, convbuf, attnf, BSD);

    // ---- FFN ----
    layernorm_kernel<<<M2, 256, 0, stream>>>(xbuf, nullptr, hbuf16, ln3_g + i * Dz, ln3_b + i * Dz, Dz, 0);
    run_gemm(stream, hbuf16, ffn1w16 + (long long)i * FFz * Dz, ffn1_b + i * FFz, nullptr, nullptr, ff116,
             M2, FFz, Dz, Dz, Dz, FFz, 1, 0, 0, 0, 1, 1.0f);
    run_gemm(stream, ff116, ffn2w16 + (long long)i * Dz * FFz, ffn2_b + i * Dz, xbuf, xbuf, nullptr,
             M2, Dz, FFz, FFz, FFz, Dz, 1, 0, 0, 0, 0, 1.0f);
  }

  // ---- final LN, dyn ----
  layernorm_kernel<<<M2, 256, 0, stream>>>(xbuf, x_out, xout16, on_g, on_b, Dz, 0);
  run_gemm(stream, xout16, opw16, op_b, nullptr, dyn_out, nullptr,
           M2, Dz, Dz, Dz, Dz, Dz, 1, 0, 0, 0, 0, 1.0f);

  // ---- head ----
  gather_last_kernel<<<EWG(Bz * Dz), 256, 0, stream>>>(x_out, csb, csb16, Bz * Dz);
  gather_last_kernel<<<EWG(Bz * Dz), 256, 0, stream>>>(actions, nullptr, actl16, Bz * Dz);

  run_gemm(stream, csb16, sew16, se_b, nullptr, comb, comb16,
           Bz, HIDz, Dz, Dz, Dz, 2 * HIDz, 1, 0, 0, 0, 0, 1.0f);
  run_gemm(stream, actl16, aew16, ae_b, nullptr, comb + HIDz, comb16 + HIDz,
           Bz, HIDz, Dz, Dz, Dz, 2 * HIDz, 1, 0, 0, 0, 0, 1.0f);

  run_gemm(stream, comb16, t0w16, t0_b, nullptr, h0, nullptr,
           Bz, HIDz, 2 * HIDz, 2 * HIDz, 2 * HIDz, HIDz, 1, 0, 0, 0, 0, 1.0f);
  layernorm_kernel<<<Bz, 256, 0, stream>>>(h0, nullptr, h116, t0ln_g, t0ln_b, HIDz, 1);
  run_gemm(stream, h116, t1w16, t1_b, nullptr, h2, nullptr,
           Bz, HIDz, HIDz, HIDz, HIDz, HIDz, 1, 0, 0, 0, 0, 1.0f);
  layernorm_kernel<<<Bz, 256, 0, stream>>>(h2, nullptr, h316, t1ln_g, t1ln_b, HIDz, 1);
  run_gemm(stream, h316, t2w16, t2_b, nullptr, delta_out, nullptr,
           Bz, Dz, HIDz, HIDz, HIDz, Dz, 1, 0, 0, 0, 0, 1.0f);

  uin_kernel<<<Bz, 256, 0, stream>>>(comb, ub16);
  run_gemm(stream, ub16, uncw16, unc_b, nullptr, unc_out, nullptr,
           Bz, Dz, HIDz, HIDz, HIDz, Dz, 1, 0, 0, 0, 3, 1.0f);

  add2_kernel<<<EWG(Bz * Dz), 256, 0, stream>>>(csb, delta_out, ns_out, Bz * Dz);
}
